// Model_29996051595771
// MI455X (gfx1250) — matrix operand order repaired and checked by reading, not compiled, not run on hardware
//
#include <hip/hip_runtime.h>

typedef __attribute__((ext_vector_type(16))) _Float16 v16h;
typedef __attribute__((ext_vector_type(8)))  float    v8f;

#define T_STEPS 100000
#define H   74
#define HP  80            // padded hidden
#define NL  3
#define GROWS (4*H)       // 296 gate rows
#define GP  304           // gate rows padded to 19*16
#define GTILES 19
#define KC  160           // concat K: 2*74=148 padded to 160 (5 chunks of 32)
#define NKC (KC/32)       // 5
#define NT  320           // 10 waves of 32
#define NWAVES (NT/32)

// ---- LDS layout (bytes), all chunks multiple of 16B ----
#define SZ_W    (NL*GP*KC*2)     // fp16 combined weights [l][row][k]   291840
#define SZ_B    (NL*GP*4)        // fused bias (b_ih+b_hh) fp32          3648
#define SZ_G    (GP*4)           // gate scratch fp32                    1216
#define SZ_C    (NL*HP*4)        // cell state fp32                       960
#define SZ_V    (NL*KC*2)        // per-layer staged [inp; h] fp16        960
#define SZ_TOP  (HP*4)           // top-layer h fp32                      320
#define SZ_WL   (HP*4)           // linear weights fp32                   320
#define SZ_BL   16               // linear bias
#define SMEM_BYTES (SZ_W+SZ_B+SZ_G+SZ_C+SZ_V+SZ_TOP+SZ_WL+SZ_BL)

// Fast activations: v_exp_f32 + v_rcp_f32 (1-ulp HW reciprocal) instead of
// the IEEE div expansion (div_scale/div_fmas/div_fixup) — this sits on the
// 100k-step serial dependence chain, so latency matters more than 0.5 ulp.
__device__ __forceinline__ float fast_rcp(float x)  { return __builtin_amdgcn_rcpf(x); }
__device__ __forceinline__ float sigf(float x)      { return fast_rcp(1.0f + __expf(-x)); }
__device__ __forceinline__ float tanh_fast(float x) { return 1.0f - 2.0f * fast_rcp(__expf(2.0f * x) + 1.0f); }

__global__ void __launch_bounds__(NT)
lstm_persistent(const float* __restrict__ seasons,
                const float* __restrict__ W_ih,
                const float* __restrict__ W_hh,
                const float* __restrict__ b_ih,
                const float* __restrict__ b_hh,
                const float* __restrict__ W_lin,
                const float* __restrict__ b_lin,
                float* __restrict__ out)
{
    extern __shared__ float4 smem4[];
    char* base = (char*)smem4;
    _Float16* Wlds    = (_Float16*)base; base += SZ_W;
    float*    biasLds = (float*)base;    base += SZ_B;
    float*    gates   = (float*)base;    base += SZ_G;
    float*    cstate  = (float*)base;    base += SZ_C;
    _Float16* vbuf    = (_Float16*)base; base += SZ_V;   // [NL][KC]
    float*    htop    = (float*)base;    base += SZ_TOP;
    float*    WlinL   = (float*)base;    base += SZ_WL;
    float*    blinL   = (float*)base;

    const int tid   = threadIdx.x;
    const int lane  = tid & 31;
    const int wave  = tid >> 5;
    const int halfw = lane >> 4;   // 0: lanes 0-15, 1: lanes 16-31
    const int lx    = lane & 15;

    // ---------------- one-time setup: pack weights into LDS as fp16 ----------------
    const int perL = GP * KC;
    for (int i = tid; i < NL * perL; i += NT) {
        int l = i / perL, rem = i - l * perL;
        int r = rem / KC, k = rem - r * KC;
        float v = 0.0f;
        if (r < GROWS) {
            if (k < H)          v = W_ih[(l * GROWS + r) * H + k];
            else if (k < 2 * H) v = W_hh[(l * GROWS + r) * H + (k - H)];
        }
        Wlds[i] = (_Float16)v;
    }
    for (int i = tid; i < NL * GP; i += NT) {
        int l = i / GP, r = i - l * GP;
        biasLds[i] = (r < GROWS) ? (b_ih[l * GROWS + r] + b_hh[l * GROWS + r]) : 0.0f;
    }
    for (int i = tid; i < NL * HP; i += NT) cstate[i] = 0.0f;
    for (int i = tid; i < NL * KC; i += NT) vbuf[i] = (_Float16)0.0f;
    for (int i = tid; i < HP; i += NT) { htop[i] = 0.0f; WlinL[i] = (i < H) ? W_lin[i] : 0.0f; }
    if (tid == 0) blinL[0] = b_lin[0];
    if (tid < H) vbuf[tid] = (_Float16)seasons[tid];   // stage x(0) into layer-0 input slot
    __syncthreads();

    // ---------------- sequential scan over time ----------------
    for (int t = 0; t < T_STEPS; ++t) {
        // prefetch next timestep's input; staged into vbuf[0] during layer-2 activation
        float xnext = 0.0f;
        if (tid < H && (t + 1) < T_STEPS) xnext = seasons[(t + 1) * H + tid];

        #pragma unroll
        for (int l = 0; l < NL; ++l) {
            const _Float16* vb = vbuf + l * KC;
            const _Float16* Wl = Wlds + l * perL;
            const float*    bl = biasLds + l * GP;

            // Hoisted A fragments: v as WMMA A-matrix row 0 (lanes 0/16 carry data),
            // loaded once per layer, reused by every tile.
            v16h af[NKC];
            #pragma unroll
            for (int kc = 0; kc < NKC; ++kc) {
                v16h a = {};
                if (lx == 0) {
                    const _Float16* p = vb + kc * 32 + halfw * 16;
                    #pragma unroll
                    for (int q = 0; q < 8; ++q) { a[q] = p[q]; a[8 + q] = p[8 + q]; }
                }
                af[kc] = a;
            }

            // gates = Wc[296x160] * v ; bias folded into C row M=0.
            for (int tile = wave; tile < GTILES; tile += NWAVES) {
                v8f acc = {};
                if (halfw == 0) acc[0] = bl[tile * 16 + lx];
                #pragma unroll
                for (int kc = 0; kc < NKC; ++kc) {
                    // B fragment: lane n holds Wc[tile*16+n][kb .. kb+15] (contiguous LDS)
                    v16h bfrag = *(const v16h*)(Wl + (tile * 16 + lx) * KC + kc * 32 + halfw * 16);
                    acc = __builtin_amdgcn_wmma_f32_16x16x32_f16(
                        false, af[kc], false, bfrag, (short)0, acc, false, false);
                }
                if (halfw == 0) gates[tile * 16 + lx] = acc[0];   // D row M=0
            }
            __syncthreads();

            // LSTM cell update (gate order i,f,g,o; bias already included).
            // Writes land directly in every consumer's staging slot.
            if (tid < H) {
                float gi = sigf(gates[tid]);
                float gf = sigf(gates[H + tid]);
                float gg = tanh_fast(gates[2 * H + tid]);
                float go = sigf(gates[3 * H + tid]);
                float c  = gf * cstate[l * HP + tid] + gi * gg;
                cstate[l * HP + tid] = c;
                float h  = go * tanh_fast(c);
                _Float16 h16 = (_Float16)h;
                vbuf[l * KC + H + tid] = h16;              // recurrent state for t+1
                if (l < NL - 1) {
                    vbuf[(l + 1) * KC + tid] = h16;        // input to next layer
                } else {
                    htop[tid] = h;                          // for output head
                    vbuf[tid] = (_Float16)xnext;            // stage x(t+1) for layer 0
                }
            }
            __syncthreads();
        }

        // output head: relu(W_lin . h_top + b_lin), wave 0 only
        if (wave == 0) {
            float p = 0.0f;
            for (int j = lane; j < H; j += 32) p += WlinL[j] * htop[j];
            p += __shfl_xor(p, 16, 32);
            p += __shfl_xor(p, 8, 32);
            p += __shfl_xor(p, 4, 32);
            p += __shfl_xor(p, 2, 32);
            p += __shfl_xor(p, 1, 32);
            if (lane == 0) out[t] = fmaxf(p + blinL[0], 0.0f);
        }
    }
}

extern "C" void kernel_launch(void* const* d_in, const int* in_sizes, int n_in,
                              void* d_out, int out_size, void* d_ws, size_t ws_size,
                              hipStream_t stream) {
    (void)in_sizes; (void)n_in; (void)d_ws; (void)ws_size; (void)out_size;
    const float* seasons = (const float*)d_in[0];
    const float* W_ih    = (const float*)d_in[1];
    const float* W_hh    = (const float*)d_in[2];
    const float* b_ih    = (const float*)d_in[3];
    const float* b_hh    = (const float*)d_in[4];
    const float* W_lin   = (const float*)d_in[5];
    const float* b_lin   = (const float*)d_in[6];
    float* out = (float*)d_out;

    lstm_persistent<<<1, NT, SMEM_BYTES, stream>>>(
        seasons, W_ih, W_hh, b_ih, b_hh, W_lin, b_lin, out);
}
